// stCluster_27092653703704
// MI455X (gfx1250) — matrix operand order repaired and checked by reading, not compiled, not run on hardware
//
#include <hip/hip_runtime.h>
#include <hip/hip_bf16.h>
#include <math.h>

// ---------------- types for CDNA5 WMMA / TDM ----------------
typedef __bf16 bf16_t;
typedef bf16_t v16bf __attribute__((ext_vector_type(16)));
typedef float  v8f   __attribute__((ext_vector_type(8)));
typedef unsigned int u32x4 __attribute__((ext_vector_type(4)));
typedef int i32x4 __attribute__((ext_vector_type(4)));
typedef int i32x8 __attribute__((ext_vector_type(8)));

#if defined(__has_builtin)
#if __has_builtin(__builtin_amdgcn_tensor_load_to_lds)
#define HAVE_TDM 1
#endif
#endif

#define IN_F   512
#define HID_F  128
#define OUT_F  64
#define K_CL   20
#define NEG_SLOPE 0.2f

union bfpack { unsigned int u; bf16_t h[2]; };

__device__ __forceinline__ float elu_f(float x) {
  return x > 0.f ? x : (__expf(x) - 1.f);
}

// float atomic max via sign-split int/uint ordering trick
__device__ __forceinline__ void atomic_max_f(float* addr, float val) {
  if (val >= 0.f) atomicMax((int*)addr, __float_as_int(val));
  else            atomicMin((unsigned int*)addr, __float_as_uint(val));
}

__device__ __forceinline__ v8f wmma_bf16(v16bf a, v16bf b, v8f c) {
  return __builtin_amdgcn_wmma_f32_16x16x32_bf16(
      false, a, false, b, (short)0, c, false, false);
}

// A fragment 16x32 bf16 from a row-major f32 row pointer.
// Layout: per lane the 16 K-values are one contiguous 16-float run:
//   elements 0..15 <-> K = 16*half .. +15
__device__ __forceinline__ v16bf load_a_frag(const float* rowPtr, int half) {
  const float4* p0 = (const float4*)(rowPtr + 16 * half);
  const float4* p1 = (const float4*)(rowPtr + 16 * half + 8);
  float4 a0 = p0[0], a1 = p0[1], b0 = p1[0], b1 = p1[1];
  v16bf a;
  a[0]=(bf16_t)a0.x;  a[1]=(bf16_t)a0.y;  a[2]=(bf16_t)a0.z;  a[3]=(bf16_t)a0.w;
  a[4]=(bf16_t)a1.x;  a[5]=(bf16_t)a1.y;  a[6]=(bf16_t)a1.z;  a[7]=(bf16_t)a1.w;
  a[8]=(bf16_t)b0.x;  a[9]=(bf16_t)b0.y;  a[10]=(bf16_t)b0.z; a[11]=(bf16_t)b0.w;
  a[12]=(bf16_t)b1.x; a[13]=(bf16_t)b1.y; a[14]=(bf16_t)b1.z; a[15]=(bf16_t)b1.w;
  return a;
}

// A fragment with fused elu(x + bias)
__device__ __forceinline__ v16bf load_a_frag_elu(const float* rowPtr,
                                                 const float* biasPtr, int half) {
  const float4* p0 = (const float4*)(rowPtr + 16 * half);
  const float4* p1 = (const float4*)(rowPtr + 16 * half + 8);
  const float4* q0 = (const float4*)(biasPtr + 16 * half);
  const float4* q1 = (const float4*)(biasPtr + 16 * half + 8);
  float4 a0 = p0[0], a1 = p0[1], b0 = p1[0], b1 = p1[1];
  float4 c0 = q0[0], c1 = q0[1], d0 = q1[0], d1 = q1[1];
  v16bf a;
  a[0]=(bf16_t)elu_f(a0.x+c0.x);  a[1]=(bf16_t)elu_f(a0.y+c0.y);
  a[2]=(bf16_t)elu_f(a0.z+c0.z);  a[3]=(bf16_t)elu_f(a0.w+c0.w);
  a[4]=(bf16_t)elu_f(a1.x+c1.x);  a[5]=(bf16_t)elu_f(a1.y+c1.y);
  a[6]=(bf16_t)elu_f(a1.z+c1.z);  a[7]=(bf16_t)elu_f(a1.w+c1.w);
  a[8]=(bf16_t)elu_f(b0.x+d0.x);  a[9]=(bf16_t)elu_f(b0.y+d0.y);
  a[10]=(bf16_t)elu_f(b0.z+d0.z); a[11]=(bf16_t)elu_f(b0.w+d0.w);
  a[12]=(bf16_t)elu_f(b1.x+d1.x); a[13]=(bf16_t)elu_f(b1.y+d1.y);
  a[14]=(bf16_t)elu_f(b1.z+d1.z); a[15]=(bf16_t)elu_f(b1.w+d1.w);
  return a;
}

// B fragment from FRAGMENT-MAJOR LDS: lane's 16 bf16 are contiguous
// (32B aligned -> 2x ds_load_b128). Element j <-> K = 16*half + j.
__device__ __forceinline__ v16bf load_b_frag_fm(const bf16_t* Bs, int t, int lane) {
  return *(const v16bf*)(Bs + ((t * 32 + lane) << 4));
}

// B fragment from row-major LDS slab Bs[32][cols] (used after TDM load).
__device__ __forceinline__ v16bf load_b_frag_rm(const bf16_t* Bs, int cols,
                                                int cb, int n, int half) {
  v16bf b;
#pragma unroll
  for (int v = 0; v < 8; ++v) {
    int k = 2 * v + 16 * half;
    b[2 * v]     = Bs[k * cols + cb + n];
    b[2 * v + 1] = Bs[(k + 1) * cols + cb + n];
  }
  return b;
}

// ---------------- 0) W_dec -> bf16 (feeds the TDM DMA) ----------------
__global__ void __launch_bounds__(256) convert_bf16(const float* __restrict__ w,
                                                    bf16_t* __restrict__ o, int n) {
  int i = blockIdx.x * blockDim.x + threadIdx.x;
  if (i < n) o[i] = (bf16_t)w[i];
}

// ---------------- 1) init scratch ----------------
__global__ void __launch_bounds__(256) init_buffers(float* __restrict__ agg,
                                                    float* __restrict__ mx,
                                                    float* __restrict__ denom,
                                                    int n) {
  long i = (long)blockIdx.x * blockDim.x + threadIdx.x;
  long stride = (long)gridDim.x * blockDim.x;
  long total = (long)n * HID_F;
  for (long j = i; j < total; j += stride) agg[j] = 0.f;
  for (long j = i; j < n; j += stride) { mx[j] = -__builtin_inff(); denom[j] = 0.f; }
}

// ---------------- 2) h = x @ W_gat  (N x 512) @ (512 x 128) ----------------
__global__ void __launch_bounds__(256) gat_gemm1(const float* __restrict__ x,
                                                 const float* __restrict__ Wg,
                                                 float* __restrict__ h, int n) {
  __shared__ __attribute__((aligned(32))) bf16_t Bs[32 * HID_F];  // fragment-major
  const int lane = threadIdx.x & 31, wave = threadIdx.x >> 5;
  const int m = lane & 15, half = lane >> 4;
  const int row0 = blockIdx.x * 128 + wave * 16;
  int row = row0 + m;
  int rc = row < n ? row : n - 1;
  const float* xr = x + (long)rc * IN_F;

  v8f acc[8] = {};
  for (int k0 = 0; k0 < IN_F; k0 += 32) {
    __syncthreads();
    // stage k-slab of W_gat as bf16 pairs directly into fragment-major layout
    for (int i = threadIdx.x; i < 16 * HID_F; i += 256) {
      int kp = i >> 7, c = i & 127;
      int k = kp * 2;
      bfpack p;
      p.h[0] = (bf16_t)Wg[(long)(k0 + k) * HID_F + c];
      p.h[1] = (bf16_t)Wg[(long)(k0 + k + 1) * HID_F + c];
      int d = (((c >> 4) * 32 + (k >> 4) * 16 + (c & 15)) << 4) + (k & 15);
      *(unsigned int*)(Bs + d) = p.u;
    }
    __syncthreads();
    v16bf a = load_a_frag(xr + k0, half);
#pragma unroll
    for (int t = 0; t < 8; ++t)
      acc[t] = wmma_bf16(a, load_b_frag_fm(Bs, t, lane), acc[t]);
  }
  if (row0 + 16 <= n) {  // interior: unguarded stores
#pragma unroll
    for (int t = 0; t < 8; ++t)
#pragma unroll
      for (int r = 0; r < 8; ++r)
        h[(long)(row0 + r + 8 * half) * HID_F + t * 16 + m] = acc[t][r];
  } else {
#pragma unroll
    for (int t = 0; t < 8; ++t)
#pragma unroll
      for (int r = 0; r < 8; ++r) {
        int orow = row0 + r + 8 * half;
        if (orow < n) h[(long)orow * HID_F + t * 16 + m] = acc[t][r];
      }
  }
}

// ---------------- 3) el/er = h @ attn_{l,r}  (wave-per-node) ----------------
__global__ void __launch_bounds__(256) attn_proj(const float* __restrict__ h,
                                                 const float* __restrict__ al,
                                                 const float* __restrict__ ar,
                                                 float* __restrict__ el,
                                                 float* __restrict__ er, int n) {
  int gw = (int)(((long)blockIdx.x * blockDim.x + threadIdx.x) >> 5);
  int lane = threadIdx.x & 31;
  if (gw >= n) return;
  float4 hv = ((const float4*)(h + (long)gw * HID_F))[lane];
  float4 lv = ((const float4*)al)[lane];
  float4 rv = ((const float4*)ar)[lane];
  float sl = hv.x * lv.x + hv.y * lv.y + hv.z * lv.z + hv.w * lv.w;
  float sr = hv.x * rv.x + hv.y * rv.y + hv.z * rv.z + hv.w * rv.w;
#pragma unroll
  for (int off = 16; off; off >>= 1) {
    sl += __shfl_xor(sl, off, 32);
    sr += __shfl_xor(sr, off, 32);
  }
  if (lane == 0) { el[gw] = sl; er[gw] = sr; }
}

// ---------------- 4) edge logits + per-dst max ----------------
__global__ void __launch_bounds__(256) edge_logits(const int* __restrict__ src,
                                                   const int* __restrict__ dst,
                                                   const float* __restrict__ el,
                                                   const float* __restrict__ er,
                                                   float* __restrict__ ebuf,
                                                   float* __restrict__ mx, int E) {
  int i = blockIdx.x * blockDim.x + threadIdx.x;
  if (i >= E) return;
  float e = el[src[i]] + er[dst[i]];
  e = e > 0.f ? e : NEG_SLOPE * e;
  ebuf[i] = e;
  atomic_max_f(&mx[dst[i]], e);
}

// ---------------- 5) a = exp(e - m[dst]); denom += a (in-place) ------------
__global__ void __launch_bounds__(256) edge_softmax_num(const int* __restrict__ dst,
                                                        float* __restrict__ ebuf,
                                                        const float* __restrict__ mx,
                                                        float* __restrict__ denom,
                                                        int E) {
  int i = blockIdx.x * blockDim.x + threadIdx.x;
  if (i >= E) return;
  int d = dst[i];
  float a = __expf(ebuf[i] - mx[d]);
  ebuf[i] = a;
  atomicAdd(&denom[d], a);
}

// ---------------- 6) agg[dst] += alpha * h[src]  (wave-per-edge) -----------
__global__ void __launch_bounds__(256) edge_scatter(const int* __restrict__ src,
                                                    const int* __restrict__ dst,
                                                    const float* __restrict__ ebuf,
                                                    const float* __restrict__ denom,
                                                    const float* __restrict__ h,
                                                    float* __restrict__ agg, int E) {
  int e = (int)(((long)blockIdx.x * blockDim.x + threadIdx.x) >> 5);
  int lane = threadIdx.x & 31;
  if (e >= E) return;
  // prefetch a future edge's gather row to hide random-access latency
  if (lane == 0 && e + 8 < E)
    __builtin_prefetch(h + (long)src[e + 8] * HID_F, 0, 0);
  int s = src[e], d = dst[e];
  float alpha = ebuf[e] / denom[d];
  float4 hv = ((const float4*)(h + (long)s * HID_F))[lane];  // 32 lanes x 4 = 128
  float* ag = agg + (long)d * HID_F + lane * 4;
  atomicAdd(ag + 0, alpha * hv.x);
  atomicAdd(ag + 1, alpha * hv.y);
  atomicAdd(ag + 2, alpha * hv.z);
  atomicAdd(ag + 3, alpha * hv.w);
}

// ---------------- 7) embed = elu(agg + bias) @ W_enc  (128 -> 64) ----------
__global__ void __launch_bounds__(256) encode_gemm(const float* __restrict__ agg,
                                                   const float* __restrict__ bias,
                                                   const float* __restrict__ Wenc,
                                                   float* __restrict__ embed, int n) {
  __shared__ __attribute__((aligned(32))) bf16_t Bs[32 * OUT_F];  // fragment-major
  const int lane = threadIdx.x & 31, wave = threadIdx.x >> 5;
  const int m = lane & 15, half = lane >> 4;
  const int row0 = blockIdx.x * 128 + wave * 16;
  int row = row0 + m;
  int rc = row < n ? row : n - 1;
  const float* ag = agg + (long)rc * HID_F;

  v8f acc[4] = {};
  for (int k0 = 0; k0 < HID_F; k0 += 32) {
    __syncthreads();
    for (int i = threadIdx.x; i < 16 * OUT_F; i += 256) {
      int kp = i >> 6, c = i & 63;
      int k = kp * 2;
      bfpack p;
      p.h[0] = (bf16_t)Wenc[(long)(k0 + k) * OUT_F + c];
      p.h[1] = (bf16_t)Wenc[(long)(k0 + k + 1) * OUT_F + c];
      int d = (((c >> 4) * 32 + (k >> 4) * 16 + (c & 15)) << 4) + (k & 15);
      *(unsigned int*)(Bs + d) = p.u;
    }
    __syncthreads();
    v16bf a = load_a_frag_elu(ag + k0, bias + k0, half);
#pragma unroll
    for (int t = 0; t < 4; ++t)
      acc[t] = wmma_bf16(a, load_b_frag_fm(Bs, t, lane), acc[t]);
  }
  if (row0 + 16 <= n) {
#pragma unroll
    for (int t = 0; t < 4; ++t)
#pragma unroll
      for (int r = 0; r < 8; ++r)
        embed[(long)(row0 + r + 8 * half) * OUT_F + t * 16 + m] = acc[t][r];
  } else {
#pragma unroll
    for (int t = 0; t < 4; ++t)
#pragma unroll
      for (int r = 0; r < 8; ++r) {
        int orow = row0 + r + 8 * half;
        if (orow < n) embed[(long)orow * OUT_F + t * 16 + m] = acc[t][r];
      }
  }
}

// ---------------- 8) pred = elu(embed @ W_dec)  (64 -> 512) ----------------
// W_dec slab (32 x 512 bf16, 32 KB) pulled into LDS by the Tensor Data Mover.
__global__ void __launch_bounds__(256) decode_gemm(const float* __restrict__ embed,
                                                   const bf16_t* __restrict__ WdecBf,
                                                   float* __restrict__ pred, int n) {
  __shared__ __attribute__((aligned(128))) bf16_t Bs[32 * IN_F];  // row-major slab
  const int lane = threadIdx.x & 31, wave = threadIdx.x >> 5;
  const int m = lane & 15, half = lane >> 4;
  const int rt = wave & 1, cg = wave >> 1;  // row tile, column group
  const int row0 = blockIdx.x * 32 + rt * 16;
  int row = row0 + m;
  int rc = row < n ? row : n - 1;
  const float* ep = embed + (long)rc * OUT_F;

  v8f acc[8] = {};
  for (int k0 = 0; k0 < OUT_F; k0 += 32) {
    __syncthreads();
#if defined(HAVE_TDM)
    if (threadIdx.x < 32) {  // wave 0 issues the DMA descriptor
      unsigned long gaddr = (unsigned long)(unsigned long long)(uintptr_t)
                            (WdecBf + (size_t)k0 * IN_F);
      unsigned lds = (unsigned)(unsigned long long)(uintptr_t)&Bs[0];
      u32x4 g0;
      g0[0] = 1u;                                    // count=1, user D#
      g0[1] = lds;                                   // lds_addr (bytes)
      g0[2] = (unsigned)(gaddr & 0xFFFFFFFFu);       // global_addr lo
      g0[3] = (unsigned)((gaddr >> 32) & 0x1FFFFFFu) // global_addr hi
              | (2u << 30);                          // type = 2 (image)
      i32x8 g1;
      g1[0] = 0x00010000;          // data_size = 1 (2 bytes/elt)
      g1[1] = (int)(512u << 16);   // tensor_dim0 = 512 (bits 63:48)
      g1[2] = (int)(64u << 16);    // tensor_dim1 = 64  (bits 95:80)
      g1[3] = (int)(512u << 16);   // tile_dim0 = 512   (bits 127:112)
      g1[4] = 32;                  // tile_dim1 = 32 rows; tile_dim2 = 0
      g1[5] = 512;                 // tensor_dim0_stride = 512
      g1[6] = 0;                   // tensor_dim1_stride = 0 (2D)
      g1[7] = 0;
      i32x4 gz = {0, 0, 0, 0};
#if __clang_major__ >= 23
      i32x8 gz8 = {0, 0, 0, 0, 0, 0, 0, 0};
      __builtin_amdgcn_tensor_load_to_lds(g0, g1, gz, gz, gz8, 0);
#else
      __builtin_amdgcn_tensor_load_to_lds(g0, g1, gz, gz, 0);
#endif
      __builtin_amdgcn_s_wait_tensorcnt(0);
    }
#else
    for (int i = threadIdx.x; i < 32 * IN_F; i += 256)
      Bs[i] = WdecBf[(long)(k0 + (i >> 9)) * IN_F + (i & 511)];
#endif
    __syncthreads();
    v16bf a = load_a_frag(ep + k0, half);
#pragma unroll
    for (int t = 0; t < 8; ++t)
      acc[t] = wmma_bf16(a, load_b_frag_rm(Bs, IN_F, cg * 128 + t * 16, m, half),
                         acc[t]);
  }
  if (row0 + 16 <= n) {  // non-temporal: pred (205 MB) is written once, not re-read
#pragma unroll
    for (int t = 0; t < 8; ++t)
#pragma unroll
      for (int r = 0; r < 8; ++r)
        __builtin_nontemporal_store(
            elu_f(acc[t][r]),
            &pred[(long)(row0 + r + 8 * half) * IN_F + cg * 128 + t * 16 + m]);
  } else {
#pragma unroll
    for (int t = 0; t < 8; ++t)
#pragma unroll
      for (int r = 0; r < 8; ++r) {
        int orow = row0 + r + 8 * half;
        if (orow < n)
          __builtin_nontemporal_store(
              elu_f(acc[t][r]),
              &pred[(long)orow * IN_F + cg * 128 + t * 16 + m]);
      }
  }
}

// ---------------- 9) DEC soft assignment (alpha=1 => exponent 1) -----------
__global__ void __launch_bounds__(256) dec_q(const float* __restrict__ embed,
                                             const float* __restrict__ centroids,
                                             float* __restrict__ q, int n) {
  __shared__ float Cs[K_CL * OUT_F];
  for (int i = threadIdx.x; i < K_CL * OUT_F; i += blockDim.x) Cs[i] = centroids[i];
  __syncthreads();
  int i = blockIdx.x * blockDim.x + threadIdx.x;
  if (i >= n) return;
  float e[OUT_F];
  const float* ep = embed + (long)i * OUT_F;
#pragma unroll
  for (int j = 0; j < OUT_F; ++j) e[j] = ep[j];
  float qk[K_CL];
  float s = 0.f;
  for (int k = 0; k < K_CL; ++k) {
    float d2 = 0.f;
#pragma unroll
    for (int j = 0; j < OUT_F; ++j) {
      float dd = e[j] - Cs[k * OUT_F + j];
      d2 += dd * dd;
    }
    float v = 1.f / (1.f + d2 + 1e-8f);
    qk[k] = v;
    s += v;
  }
  float inv = 1.f / s;
  for (int k = 0; k < K_CL; ++k) q[(long)i * K_CL + k] = qk[k] * inv;
}

// ---------------- launch ----------------
extern "C" void kernel_launch(void* const* d_in, const int* in_sizes, int n_in,
                              void* d_out, int out_size, void* d_ws, size_t ws_size,
                              hipStream_t stream) {
  const float* x    = (const float*)d_in[0];
  const float* Wg   = (const float*)d_in[1];
  const float* al   = (const float*)d_in[2];
  const float* ar   = (const float*)d_in[3];
  const float* bias = (const float*)d_in[4];
  const float* Wenc = (const float*)d_in[5];
  const float* Wdec = (const float*)d_in[6];
  const float* cent = (const float*)d_in[7];
  const int*   src  = (const int*)d_in[8];
  const int*   dst  = (const int*)d_in[9];
  const int n = in_sizes[0] / IN_F;   // 100000
  const int E = in_sizes[8];          // 900000

  // workspace carve-out (floats)
  float* ws    = (float*)d_ws;
  size_t off   = 0;
  float* h     = ws + off; off += (size_t)n * HID_F;
  float* agg   = ws + off; off += (size_t)n * HID_F;
  float* el    = ws + off; off += (size_t)n;
  float* er    = ws + off; off += (size_t)n;
  float* mx    = ws + off; off += (size_t)n;
  float* denom = ws + off; off += (size_t)n;
  float* ebuf  = ws + off; off += (size_t)E;
  bf16_t* wdecBf = (bf16_t*)(ws + off); off += (size_t)(OUT_F * IN_F) / 2;

  // outputs: embed (N*64) | pred_gene (N*512) | q (N*20)
  float* embed = (float*)d_out;
  float* pred  = embed + (size_t)n * OUT_F;
  float* q     = pred + (size_t)n * IN_F;

  const int blk128 = (n + 127) / 128;

  convert_bf16<<<(OUT_F * IN_F + 255) / 256, 256, 0, stream>>>(Wdec, wdecBf,
                                                               OUT_F * IN_F);
  init_buffers<<<512, 256, 0, stream>>>(agg, mx, denom, n);
  gat_gemm1<<<blk128, 256, 0, stream>>>(x, Wg, h, n);
  attn_proj<<<(n * 32 + 255) / 256, 256, 0, stream>>>(h, al, ar, el, er, n);
  edge_logits<<<(E + 255) / 256, 256, 0, stream>>>(src, dst, el, er, ebuf, mx, E);
  edge_softmax_num<<<(E + 255) / 256, 256, 0, stream>>>(dst, ebuf, mx, denom, E);
  edge_scatter<<<(E * 32 + 255) / 256, 256, 0, stream>>>(src, dst, ebuf, denom, h,
                                                         agg, E);
  encode_gemm<<<blk128, 256, 0, stream>>>(agg, bias, Wenc, embed, n);
  decode_gemm<<<(n + 31) / 32, 256, 0, stream>>>(embed, wdecBf, pred, n);
  dec_q<<<(n + 255) / 256, 256, 0, stream>>>(embed, cent, q, n);
}
